// ARMAAttentionLSTM_56564719288947
// MI455X (gfx1250) — matrix operand order repaired and checked by reading, not compiled, not run on hardware
//
#include <hip/hip_runtime.h>
#include <stdint.h>

typedef __bf16 bf16_t;
typedef __attribute__((ext_vector_type(16))) __bf16 v16bf;
typedef __attribute__((ext_vector_type(8)))  float  v8f;

#define BB   16
#define TT   1024
#define FIN  128
#define HH   256
#define MM   (BB*TT)     // 16384

union BF16x16 { v16bf v; uint4 q[2]; };

__device__ __forceinline__ v8f wmma_bf16(v16bf a, v16bf b, v8f c) {
  return __builtin_amdgcn_wmma_f32_16x16x32_bf16(false, a, false, b, (short)0, c, false, false);
}

// ---------------------------------------------------------------------------
// Async global->LDS copy (16B per call), guarded so compile never breaks.
#if __has_builtin(__builtin_amdgcn_global_load_async_to_lds_b128)
#define HAVE_ASYNC_LDS 1
typedef int v4i_vs __attribute__((vector_size(16)));
typedef __attribute__((address_space(1))) v4i_vs* gptr_v4i;
typedef __attribute__((address_space(3))) v4i_vs* lptr_v4i;
__device__ __forceinline__ void async_cp16(const void* g, void* l) {
  __builtin_amdgcn_global_load_async_to_lds_b128((gptr_v4i)(g), (lptr_v4i)(l), 0, 0);
}
__device__ __forceinline__ void async_wait() {
#if __has_builtin(__builtin_amdgcn_s_wait_asynccnt)
  __builtin_amdgcn_s_wait_asynccnt(0);
#else
  asm volatile("s_wait_asynccnt 0x0" ::: "memory");
#endif
}
#else
#define HAVE_ASYNC_LDS 0
__device__ __forceinline__ void async_cp16(const void* g, void* l) {
  *(uint4*)l = *(const uint4*)g;
}
__device__ __forceinline__ void async_wait() {}
#endif

// A-fragment (16x32 bf16, M x K) from row-major [.., lda] bf16.
__device__ __forceinline__ v16bf load_a_frag(const bf16_t* base, int row0, int lda, int k0, int lane) {
  int m   = lane & 15;
  int off = (lane < 16) ? 0 : 8;
  const bf16_t* p = base + (size_t)(row0 + m) * lda + k0 + off;
  BF16x16 u;
  u.q[0] = *(const uint4*)(p);
  u.q[1] = *(const uint4*)(p + 16);
  return u.v;
}

// B-fragment (32x16 bf16, K x N) where memory holds W[N][K] row-major (K contiguous).
// Works for global or LDS base pointers (generic addressing).
__device__ __forceinline__ v16bf load_b_frag(const bf16_t* base, int ncol0, int ldb, int k0, int lane) {
  int n   = lane & 15;
  int off = (lane < 16) ? 0 : 8;
  const bf16_t* p = base + (size_t)(ncol0 + n) * ldb + k0 + off;
  BF16x16 u;
  u.q[0] = *(const uint4*)(p);
  u.q[1] = *(const uint4*)(p + 16);
  return u.v;
}

// ---------------------------------------------------------------------------
__global__ void k_f32_to_bf16(const float* __restrict__ in, bf16_t* __restrict__ out, size_t n) {
  size_t i = (size_t)blockIdx.x * blockDim.x + threadIdx.x;
  if (i < n) out[i] = (bf16_t)in[i];
}

// ---------------------------------------------------------------------------
// bf16 WMMA GEMM: C[M,N] = A[M,K] * W[N,K]^T + bias0 + bias1 (+residual).
// The 64xK W tile is staged once per workgroup into LDS via async copies;
// every wave then reads B fragments with ds_load_b128. Each wave computes a
// 16(M) x 64(N) strip; block = 8 waves -> 128(M) x 64(N). K <= 256.
__global__ __launch_bounds__(256) void k_gemm_bf16(
    const bf16_t* __restrict__ A, const bf16_t* __restrict__ W,
    const float* __restrict__ bias0, const float* __restrict__ bias1,
    const float* __restrict__ residual,
    float* __restrict__ outF, bf16_t* __restrict__ outB,
    int M, int N, int K) {
  __shared__ __align__(16) bf16_t wtile[64 * 256];   // up to 32 KB

  int tid  = threadIdx.x;
  int lane = tid & 31;
  int wv   = tid >> 5;
  int row0 = (blockIdx.y * 8 + wv) * 16;
  int nc0  = blockIdx.x * 64;

  // stage W[nc0 .. nc0+64) x K : contiguous 64*K*2 bytes
  {
    const uint8_t* gsrc = (const uint8_t*)(W + (size_t)nc0 * K);
    uint8_t* ldst = (uint8_t*)wtile;
    int bytes = 64 * K * 2;
    for (int ofs = tid * 16; ofs < bytes; ofs += 256 * 16)
      async_cp16(gsrc + ofs, ldst + ofs);
  }
  async_wait();
  __syncthreads();

  v8f acc[4] = {};
  for (int k0 = 0; k0 < K; k0 += 32) {
    v16bf a = load_a_frag(A, row0, K, k0, lane);
#pragma unroll
    for (int j = 0; j < 4; ++j) {
      v16bf b = load_b_frag(wtile, j * 16, K, k0, lane);
      acc[j] = wmma_bf16(a, b, acc[j]);
    }
  }
  int rb = (lane & 16) ? 8 : 0;
  int cl = lane & 15;
#pragma unroll
  for (int j = 0; j < 4; ++j) {
    int col = nc0 + j * 16 + cl;
    float bv = (bias0 ? bias0[col] : 0.f) + (bias1 ? bias1[col] : 0.f);
#pragma unroll
    for (int r = 0; r < 8; ++r) {
      int row = row0 + rb + r;
      float v = acc[j][r] + bv;
      size_t idx = (size_t)row * N + col;
      if (residual) v += residual[idx];
      if (outF) outF[idx] = v;
      if (outB) outB[idx] = (bf16_t)v;
    }
  }
}

// ---------------------------------------------------------------------------
// Persistent single-workgroup LSTM scan. Per step t:
// gates[16][1024] = xg[:,t,:] + h @ Whh^T (WMMA), then elementwise cell update
// with h (bf16) / c (f32) resident in LDS. xg is bf16 to halve HBM traffic.
__global__ __launch_bounds__(1024) void k_lstm_scan(
    const bf16_t* __restrict__ xg,      // [B, T, 4H] bf16
    const bf16_t* __restrict__ Whh,     // [4H, H] bf16 row-major
    float* __restrict__ outF,           // [B, T, H] or null
    bf16_t* __restrict__ outB,          // [B, T, H]
    float* __restrict__ hn, float* __restrict__ cn) {
  __shared__ __align__(16) bf16_t h_lds[BB][HH];        // 8 KB
  __shared__ __align__(16) float  c_lds[BB][HH];        // 16 KB
  __shared__ __align__(16) bf16_t g_lds[BB][4 * HH];    // 32 KB

  int tid  = threadIdx.x;
  int lane = tid & 31;
  int wv   = tid >> 5;          // 0..31
  int nt0  = wv * 2;            // two 16-wide N tiles per wave -> 64 tiles

  for (int i = tid; i < BB * HH; i += 1024) {
    ((bf16_t*)h_lds)[i] = (bf16_t)0.f;
    ((float*)c_lds)[i]  = 0.f;
  }
  __syncthreads();

  int rb = (lane & 16) ? 8 : 0;
  int cl = lane & 15;

  for (int t = 0; t < TT; ++t) {
    v8f acc[2] = {};
#pragma unroll
    for (int c = 0; c < 8; ++c) {
      int m = lane & 15;
      int off = (lane < 16) ? 0 : 8;
      const bf16_t* p = &h_lds[m][c * 32 + off];
      BF16x16 ua;
      ua.q[0] = *(const uint4*)(p);
      ua.q[1] = *(const uint4*)(p + 16);
#pragma unroll
      for (int j = 0; j < 2; ++j) {
        v16bf b = load_b_frag(Whh, (nt0 + j) * 16, HH, c * 32, lane);
        acc[j] = wmma_bf16(ua.v, b, acc[j]);
      }
    }
#pragma unroll
    for (int j = 0; j < 2; ++j) {
      int col = (nt0 + j) * 16 + cl;
#pragma unroll
      for (int r = 0; r < 8; ++r) {
        int b = rb + r; // batch index
        float g = acc[j][r] + (float)xg[((size_t)b * TT + t) * (4 * HH) + col];
        g_lds[b][col] = (bf16_t)g;
      }
    }
    __syncthreads();

    for (int i = tid; i < BB * HH; i += 1024) {
      int b = i >> 8;
      int j = i & 255;
      float gi = (float)g_lds[b][j];
      float gf = (float)g_lds[b][HH + j];
      float gg = (float)g_lds[b][2 * HH + j];
      float go = (float)g_lds[b][3 * HH + j];
      float ii = 1.f / (1.f + __expf(-gi));
      float ff = 1.f / (1.f + __expf(-gf));
      float gt = tanhf(gg);
      float oo = 1.f / (1.f + __expf(-go));
      float cv = ff * c_lds[b][j] + ii * gt;
      float hv = oo * tanhf(cv);
      c_lds[b][j] = cv;
      h_lds[b][j] = (bf16_t)hv;
      size_t oidx = ((size_t)b * TT + t) * HH + j;
      outB[oidx] = (bf16_t)hv;
      if (outF) outF[oidx] = hv;
      if (t == TT - 1) {
        hn[b * HH + j] = hv;
        cn[b * HH + j] = cv;
      }
    }
    __syncthreads();
  }
}

// ---------------------------------------------------------------------------
// V [B,T,H] -> Vt [B,H,T] (bf16)
__global__ void k_transpose_v(const bf16_t* __restrict__ V, bf16_t* __restrict__ Vt) {
  size_t i = (size_t)blockIdx.x * blockDim.x + threadIdx.x;
  if (i >= (size_t)BB * TT * HH) return;
  int h = (int)(i % HH);
  size_t bt = i / HH;
  int t = (int)(bt % TT);
  int b = (int)(bt / TT);
  Vt[((size_t)b * HH + h) * TT + t] = V[i];
}

// ---------------------------------------------------------------------------
// Flash attention with ARMA banded modulation.
// Block = 8 waves = 8 consecutive 16-query blocks of ONE batch. Per 32-key
// iteration the K tile (32x256) and V^T tile (256x32) are async-staged into
// LDS once and shared by all 8 waves. acc (16x256 f32) stays in registers.
__global__ __launch_bounds__(256) void k_flash_arma(
    const bf16_t* __restrict__ Q, const bf16_t* __restrict__ Kmat,
    const bf16_t* __restrict__ Vt,
    const float* __restrict__ ar_w, const float* __restrict__ ma_w,
    bf16_t* __restrict__ attended) {
  __shared__ __align__(16) bf16_t Ktile[32 * HH];    // 16 KB
  __shared__ __align__(16) bf16_t Vtile[HH * 32];    // 16 KB
  __shared__ __align__(16) bf16_t Pbuf[8][16][32];   // 8 KB

  int tid  = threadIdx.x;
  int lane = tid & 31;
  int wv   = tid >> 5;
  int qb   = blockIdx.x * 8 + wv;        // 0 .. B*T/16-1
  int b    = qb >> 6;                    // 64 q-blocks per batch (8 | 64 -> same b per block)
  int q0   = (qb & 63) << 4;

  const bf16_t* Qb = Q    + (size_t)b * TT * HH;
  const bf16_t* Kb = Kmat + (size_t)b * TT * HH;
  const bf16_t* Vb = Vt   + (size_t)b * HH * TT;

  float a0 = ar_w[0], a1 = ar_w[1], a2 = ar_w[2];
  float m0 = ma_w[0], m1 = ma_w[1];
  float mod[4];
  mod[0] = __expf(m0 + m1);
  mod[1] = __expf(a0 + m0 + m1);
  mod[2] = __expf(a1 + m1);
  mod[3] = __expf(a2);
  const float scale = 0.0625f;  // 1/sqrt(H)

  v16bf qf[8];
#pragma unroll
  for (int c = 0; c < 8; ++c) qf[c] = load_a_frag(Qb, q0, HH, c * 32, lane);

  v8f acc[16] = {};
  float mrow[8], lrow[8];
#pragma unroll
  for (int r = 0; r < 8; ++r) { mrow[r] = -1e30f; lrow[r] = 0.f; }

  int rb = (lane & 16) ? 8 : 0;
  int cl = lane & 15;

  for (int kb = 0; kb < TT; kb += 32) {
    // ---- stage K tile (contiguous) and V^T tile (strided) into LDS ----
    {
      const uint8_t* src = (const uint8_t*)(Kb + (size_t)kb * HH);
      uint8_t* dst = (uint8_t*)Ktile;
#pragma unroll
      for (int it = 0; it < 4; ++it)
        async_cp16(src + (tid + it * 256) * 16, dst + (tid + it * 256) * 16);
#pragma unroll
      for (int it = 0; it < 4; ++it) {
        int c2 = tid + it * 256;         // 0..1023, 16B chunks
        int hrow = c2 >> 2, cp = c2 & 3;
        async_cp16(Vb + (size_t)hrow * TT + kb + cp * 8,
                   Vtile + hrow * 32 + cp * 8);
      }
    }
    if (kb + 32 < TT)
      __builtin_prefetch(Kb + (size_t)(kb + 32 + lane) * HH, 0, 1);
    async_wait();
    __syncthreads();

    // ---- S = Q K^T (B fragments from LDS) ----
    v8f s0 = {}, s1 = {};
#pragma unroll
    for (int c = 0; c < 8; ++c) {
      v16bf b0 = load_b_frag(Ktile, 0,  HH, c * 32, lane);
      v16bf b1 = load_b_frag(Ktile, 16, HH, c * 32, lane);
      s0 = wmma_bf16(qf[c], b0, s0);
      s1 = wmma_bf16(qf[c], b1, s1);
    }

    // ---- ARMA band modulation + online softmax ----
    float p0[8], p1[8], sc[8];
#pragma unroll
    for (int r = 0; r < 8; ++r) {
      int q  = q0 + rb + r;
      int k0 = kb + cl;
      int k1 = kb + 16 + cl;
      int d0 = (q > k0) ? (q - k0) : (k0 - q);
      int d1 = (q > k1) ? (q - k1) : (k1 - q);
      float v0 = s0[r] * scale * (d0 < 4 ? mod[d0] : 1.f);
      float v1 = s1[r] * scale * (d1 < 4 ? mod[d1] : 1.f);
      float mx = fmaxf(v0, v1);
#pragma unroll
      for (int o = 1; o < 16; o <<= 1) mx = fmaxf(mx, __shfl_xor(mx, o, 32));
      float mn = fmaxf(mrow[r], mx);
      sc[r] = __expf(mrow[r] - mn);
      mrow[r] = mn;
      v0 = __expf(v0 - mn);
      v1 = __expf(v1 - mn);
      p0[r] = v0; p1[r] = v1;
      float sum = v0 + v1;
#pragma unroll
      for (int o = 1; o < 16; o <<= 1) sum += __shfl_xor(sum, o, 32);
      lrow[r] = lrow[r] * sc[r] + sum;
    }
#pragma unroll
    for (int nt = 0; nt < 16; ++nt)
#pragma unroll
      for (int r = 0; r < 8; ++r) acc[nt][r] *= sc[r];

    // ---- wave-local P transpose through LDS ----
#pragma unroll
    for (int r = 0; r < 8; ++r) {
      Pbuf[wv][rb + r][cl]      = (bf16_t)p0[r];
      Pbuf[wv][rb + r][16 + cl] = (bf16_t)p1[r];
    }
    asm volatile("s_wait_dscnt 0x0" ::: "memory");
    int m = lane & 15;
    int off = (lane < 16) ? 0 : 8;
    BF16x16 up;
    up.q[0] = *(const uint4*)&Pbuf[wv][m][off];
    up.q[1] = *(const uint4*)&Pbuf[wv][m][off + 16];

    // ---- acc += P @ V (B fragments from LDS V^T tile) ----
#pragma unroll
    for (int nt = 0; nt < 16; ++nt) {
      const bf16_t* p = Vtile + (nt * 16 + cl) * 32 + ((lane & 16) ? 8 : 0);
      BF16x16 ub;
      ub.q[0] = *(const uint4*)(p);
      ub.q[1] = *(const uint4*)(p + 16);
      acc[nt] = wmma_bf16(up.v, ub.v, acc[nt]);
    }
    __syncthreads();   // all waves done with K/V tiles before next stage
  }

#pragma unroll
  for (int nt = 0; nt < 16; ++nt) {
#pragma unroll
    for (int r = 0; r < 8; ++r) {
      float v = acc[nt][r] / lrow[r];
      size_t idx = ((size_t)b * TT + q0 + rb + r) * HH + nt * 16 + cl;
      attended[idx] = (bf16_t)v;
    }
  }
}

// ---------------------------------------------------------------------------
extern "C" void kernel_launch(void* const* d_in, const int* in_sizes, int n_in,
                              void* d_out, int out_size, void* d_ws, size_t ws_size,
                              hipStream_t stream) {
  const float* x    = (const float*)d_in[0];
  const float* Wih0 = (const float*)d_in[1];
  const float* Whh0 = (const float*)d_in[2];
  const float* bih0 = (const float*)d_in[3];
  const float* bhh0 = (const float*)d_in[4];
  const float* Wih1 = (const float*)d_in[5];
  const float* Whh1 = (const float*)d_in[6];
  const float* bih1 = (const float*)d_in[7];
  const float* bhh1 = (const float*)d_in[8];
  const float* ar_w = (const float*)d_in[9];
  const float* ma_w = (const float*)d_in[10];
  const float* Wq = (const float*)d_in[11];
  const float* bq = (const float*)d_in[12];
  const float* Wk = (const float*)d_in[13];
  const float* bk = (const float*)d_in[14];
  const float* Wv = (const float*)d_in[15];
  const float* bv = (const float*)d_in[16];
  const float* Wo = (const float*)d_in[17];
  const float* bo = (const float*)d_in[18];

  float* out = (float*)d_out;                 // [B,T,H]
  float* hn0 = out + (size_t)MM * HH;         // h_n[0]
  float* hn1 = hn0 + (size_t)BB * HH;         // h_n[1]
  float* cn0 = hn1 + (size_t)BB * HH;         // c_n[0]
  float* cn1 = cn0 + (size_t)BB * HH;         // c_n[1]

  uint8_t* ws = (uint8_t*)d_ws;
  size_t off = 0;
  auto alloc = [&](size_t bytes) { uint8_t* p = ws + off; off += (bytes + 255) & ~(size_t)255; return p; };

  bf16_t* xg      = (bf16_t*)alloc((size_t)MM * 4 * HH * sizeof(bf16_t)); // reused for both layers
  bf16_t* x_bf    = (bf16_t*)alloc((size_t)MM * FIN * sizeof(bf16_t));
  bf16_t* h1_bf   = (bf16_t*)alloc((size_t)MM * HH * sizeof(bf16_t));
  float*  lout_f  = (float*)alloc((size_t)MM * HH * sizeof(float));
  bf16_t* lout_bf = (bf16_t*)alloc((size_t)MM * HH * sizeof(bf16_t));
  bf16_t* Qb      = (bf16_t*)alloc((size_t)MM * HH * sizeof(bf16_t));
  bf16_t* Kb      = (bf16_t*)alloc((size_t)MM * HH * sizeof(bf16_t));
  bf16_t* Vb      = (bf16_t*)alloc((size_t)MM * HH * sizeof(bf16_t));
  bf16_t* Vt      = (bf16_t*)alloc((size_t)MM * HH * sizeof(bf16_t));
  bf16_t* att_bf  = (bf16_t*)alloc((size_t)MM * HH * sizeof(bf16_t));
  bf16_t* Wih0b = (bf16_t*)alloc((size_t)4 * HH * FIN * sizeof(bf16_t));
  bf16_t* Whh0b = (bf16_t*)alloc((size_t)4 * HH * HH * sizeof(bf16_t));
  bf16_t* Wih1b = (bf16_t*)alloc((size_t)4 * HH * HH * sizeof(bf16_t));
  bf16_t* Whh1b = (bf16_t*)alloc((size_t)4 * HH * HH * sizeof(bf16_t));
  bf16_t* Wqb   = (bf16_t*)alloc((size_t)HH * HH * sizeof(bf16_t));
  bf16_t* Wkb   = (bf16_t*)alloc((size_t)HH * HH * sizeof(bf16_t));
  bf16_t* Wvb   = (bf16_t*)alloc((size_t)HH * HH * sizeof(bf16_t));
  bf16_t* Wob   = (bf16_t*)alloc((size_t)HH * HH * sizeof(bf16_t));

  auto cvt = [&](const float* src, bf16_t* dst, size_t n) {
    k_f32_to_bf16<<<(unsigned)((n + 255) / 256), 256, 0, stream>>>(src, dst, n);
  };
  cvt(x, x_bf, (size_t)MM * FIN);
  cvt(Wih0, Wih0b, (size_t)4 * HH * FIN);
  cvt(Whh0, Whh0b, (size_t)4 * HH * HH);
  cvt(Wih1, Wih1b, (size_t)4 * HH * HH);
  cvt(Whh1, Whh1b, (size_t)4 * HH * HH);
  cvt(Wq, Wqb, (size_t)HH * HH);
  cvt(Wk, Wkb, (size_t)HH * HH);
  cvt(Wv, Wvb, (size_t)HH * HH);
  cvt(Wo, Wob, (size_t)HH * HH);

  auto gemm = [&](const bf16_t* A, const bf16_t* W, const float* b0, const float* b1,
                  const float* res, float* oF, bf16_t* oB, int M, int N, int K) {
    dim3 grid(N / 64, M / 128);
    k_gemm_bf16<<<grid, 256, 0, stream>>>(A, W, b0, b1, res, oF, oB, M, N, K);
  };

  // layer 0: xg = x @ Wih0^T + bih0 + bhh0 ; scan
  gemm(x_bf, Wih0b, bih0, bhh0, nullptr, nullptr, xg, MM, 4 * HH, FIN);
  k_lstm_scan<<<1, 1024, 0, stream>>>(xg, Whh0b, nullptr, h1_bf, hn0, cn0);

  // layer 1
  gemm(h1_bf, Wih1b, bih1, bhh1, nullptr, nullptr, xg, MM, 4 * HH, HH);
  k_lstm_scan<<<1, 1024, 0, stream>>>(xg, Whh1b, lout_f, lout_bf, hn1, cn1);

  // Q, K, V projections (bf16 outputs)
  gemm(lout_bf, Wqb, bq, nullptr, nullptr, nullptr, Qb, MM, HH, HH);
  gemm(lout_bf, Wkb, bk, nullptr, nullptr, nullptr, Kb, MM, HH, HH);
  gemm(lout_bf, Wvb, bv, nullptr, nullptr, nullptr, Vb, MM, HH, HH);

  // V transpose for P@V fragment loads
  k_transpose_v<<<(unsigned)(((size_t)MM * HH + 255) / 256), 256, 0, stream>>>(Vb, Vt);

  // flash attention with ARMA band
  k_flash_arma<<<MM / 16 / 8, 256, 0, stream>>>(Qb, Kb, Vt, ar_w, ma_w, att_bf);

  // out = attended @ Wo^T + bo + lstm_out
  gemm(att_bf, Wob, bo, nullptr, lout_f, out, nullptr, MM, HH, HH);
}
